// MultiheadSdpaDiff4_3427383902621
// MI455X (gfx1250) — matrix operand order repaired and checked by reading, not compiled, not run on hardware
//
#include <hip/hip_runtime.h>
#include <hip/hip_bf16.h>
#include <stdint.h>

// ---------------- problem constants ----------------
#define EMBED   2048
#define T_SEQ   2048
#define BATCH   2
#define BT      (BATCH * T_SEQ)        // 4096 rows
#define HD      64
#define NPAIR   16                     // H = 16 differential pairs
#define KV_COLS 512                    // k/v projection width
#define LAMBDA_INIT_F 0.78360576553f   // 0.8 - 0.6*exp(-0.3*12)

typedef unsigned short u16;
typedef __attribute__((ext_vector_type(16))) __bf16 bf16x16;
typedef __attribute__((ext_vector_type(8)))  float  f32x8;

// CDNA5 async global->LDS path (ASYNCcnt-tracked). Implemented with inline asm
// to avoid the builtin's CUDA-language-address-space parameter types, which
// cannot be spelled in HIP source. LDS byte address = low 32 bits of the flat
// pointer (ISA 10.2: LDS aperture maps addr[31:0] directly to LDS).
#if defined(__gfx1250__)
#define HAVE_ASYNC 1
#define ASYNC_WAIT() asm volatile("s_wait_asynccnt 0" ::: "memory")
static __device__ __forceinline__ void async_g2l_b128(const void* g, void* l) {
  unsigned lds = (unsigned)(unsigned long long)(uintptr_t)l;
  unsigned long long ga = (unsigned long long)(uintptr_t)g;
  asm volatile("global_load_async_to_lds_b128 %0, %1, off"
               :: "v"(lds), "v"(ga)
               : "memory");
}
#else
#define HAVE_ASYNC 0
#define ASYNC_WAIT()
#endif

struct uint4x2 { uint4 a, b; };

static __device__ __forceinline__ bf16x16 frag2(const u16* p0, const u16* p1) {
  uint4x2 u;
  u.a = *(const uint4*)p0;
  u.b = *(const uint4*)p1;
  return __builtin_bit_cast(bf16x16, u);
}

static __device__ __forceinline__ f32x8 wmma_bf16(bf16x16 a, bf16x16 b, f32x8 c) {
  // (neg_a, A, neg_b, B, c_mod, C, reuse_a, reuse_b)
  return __builtin_amdgcn_wmma_f32_16x16x32_bf16(false, a, false, b, (short)0, c,
                                                 false, false);
}

static __device__ __forceinline__ u16 f2bf(float f) {
  unsigned int u = __float_as_uint(f);
  u += 0x7FFFu + ((u >> 16) & 1u);     // round-to-nearest-even
  return (u16)(u >> 16);
}

// ---------------- fp32 -> bf16 convert ----------------
__global__ __launch_bounds__(256) void cvt_f32_bf16(const float* __restrict__ in,
                                                    u16* __restrict__ out, int n) {
  int i = (blockIdx.x * 256 + threadIdx.x) * 4;
  if (i + 3 < n) {
    float4 v = *(const float4*)(in + i);
    unsigned int a = (unsigned int)f2bf(v.x) | ((unsigned int)f2bf(v.y) << 16);
    unsigned int b = (unsigned int)f2bf(v.z) | ((unsigned int)f2bf(v.w) << 16);
    uint2 r; r.x = a; r.y = b;
    *(uint2*)(out + i) = r;
  }
}

// ---------------- lambda_full scalar ----------------
__global__ __launch_bounds__(32) void lambda_kernel(const float* __restrict__ lq1,
                                                    const float* __restrict__ lk1,
                                                    const float* __restrict__ lq2,
                                                    const float* __restrict__ lk2,
                                                    float* __restrict__ out) {
  int t = threadIdx.x;  // 32 lanes, HD=64 -> 2 elements each
  float s1 = lq1[t] * lk1[t] + lq1[t + 32] * lk1[t + 32];
  float s2 = lq2[t] * lk2[t] + lq2[t + 32] * lk2[t + 32];
  for (int m = 16; m >= 1; m >>= 1) {
    s1 += __shfl_xor(s1, m, 32);
    s2 += __shfl_xor(s2, m, 32);
  }
  if (t == 0) out[0] = expf(s1) - expf(s2) + LAMBDA_INIT_F;
}

// ---------------- V transpose: (b,t,512) -> (b,512,t) ----------------
__global__ __launch_bounds__(256) void transpose_v(const u16* __restrict__ in,
                                                   u16* __restrict__ out) {
  __shared__ u16 tile[32][33];
  int b = blockIdx.z;
  int c0 = blockIdx.x * 32, t0 = blockIdx.y * 32;
  int tx = threadIdx.x, ty = threadIdx.y;  // (32,8)
  const u16* ib = in + (size_t)b * T_SEQ * KV_COLS;
  u16* ob = out + (size_t)b * KV_COLS * T_SEQ;
  for (int i = 0; i < 4; i++)
    tile[ty + i * 8][tx] = ib[(size_t)(t0 + ty + i * 8) * KV_COLS + c0 + tx];
  __syncthreads();
  for (int i = 0; i < 4; i++)
    ob[(size_t)(c0 + ty + i * 8) * T_SEQ + t0 + tx] = tile[tx][ty + i * 8];
}

// ---------------- GEMM: C[M,N] = A[M,K] * W[N,K]^T (bf16 in, f32 acc) --------
// block = 256 threads = 8 waves; block tile 128x128; wave tile 64x32 (4x2 frags)
// Double-buffered LDS staging via GLOBAL_LOAD_ASYNC_TO_LDS_B128 (inline asm).
template <bool OUTF32>
__global__ __launch_bounds__(256) void gemm_xwT(const u16* __restrict__ A,
                                                const u16* __restrict__ W,
                                                void* __restrict__ Cp,
                                                int M, int N, int K) {
  constexpr int LDT = 40;  // 32 + 8 pad: conflict-free 16B ds loads
  __shared__ u16 As[2][128 * LDT];
  __shared__ u16 Bs[2][128 * LDT];
  const int tid = threadIdx.x;
  const int lane = tid & 31, wave = tid >> 5;
  const int wm = wave >> 2, wn = wave & 3;
  const int lrw = lane & 15, lh = lane >> 4;
  const int rowBase = blockIdx.y * 128, colBase = blockIdx.x * 128;

  f32x8 acc[4][2] = {};

  auto stage = [&](int buf, int k0) {
    for (int i = 0; i < 2; i++) {
      int lin = tid + i * 256;
      int r = lin >> 2, seg = lin & 3;
      const u16* ga = &A[(size_t)(rowBase + r) * K + k0 + seg * 8];
      const u16* gb = &W[(size_t)(colBase + r) * K + k0 + seg * 8];
#if HAVE_ASYNC
      async_g2l_b128(ga, &As[buf][r * LDT + seg * 8]);
      async_g2l_b128(gb, &Bs[buf][r * LDT + seg * 8]);
#else
      *(uint4*)&As[buf][r * LDT + seg * 8] = *(const uint4*)ga;
      *(uint4*)&Bs[buf][r * LDT + seg * 8] = *(const uint4*)gb;
#endif
    }
  };

  stage(0, 0);
  ASYNC_WAIT();
  __syncthreads();

  int cur = 0;
  for (int k0 = 0; k0 < K; k0 += 32) {
    if (k0 + 32 < K) stage(cur ^ 1, k0 + 32);  // prefetch next tile (async)

    bf16x16 af[4], bfr[2];
    for (int mf = 0; mf < 4; mf++) {
      // A-frag: lane = M row; k = lh*8 + {0..7} and +16
      const u16* p = &As[cur][(wm * 64 + mf * 16 + lrw) * LDT + lh * 8];
      af[mf] = frag2(p, p + 16);
    }
    for (int nf = 0; nf < 2; nf++) {
      // B-frag: lane = N col; k = lh*8 + {0..7} and +16
      const u16* p = &Bs[cur][(wn * 32 + nf * 16 + lrw) * LDT + lh * 8];
      bfr[nf] = frag2(p, p + 16);
    }
    for (int mf = 0; mf < 4; mf++)
      for (int nf = 0; nf < 2; nf++)
        acc[mf][nf] = wmma_bf16(af[mf], bfr[nf], acc[mf][nf]);

    ASYNC_WAIT();       // own prefetch landed in LDS
    __syncthreads();    // all waves done reading `cur`, all prefetch visible
    cur ^= 1;
  }

  // C layout: lanes 0-15: N=lane, M=j ; lanes 16-31: N=lane-16, M=j+8
  for (int mf = 0; mf < 4; mf++)
    for (int nf = 0; nf < 2; nf++)
      for (int j = 0; j < 8; j++) {
        int r = rowBase + wm * 64 + mf * 16 + j + 8 * lh;
        int c = colBase + wn * 32 + nf * 16 + lrw;
        if (OUTF32)
          ((float*)Cp)[(size_t)r * N + c] = acc[mf][nf][j];
        else
          ((u16*)Cp)[(size_t)r * N + c] = f2bf(acc[mf][nf][j]);
      }
}

// ---------------- differential flash attention ----------------
// one wave per block; block handles one (batch, head-pair, 16-row q tile)
__global__ __launch_bounds__(32) void diff_attn(const u16* __restrict__ Qm,
                                                const u16* __restrict__ Km,
                                                const u16* __restrict__ Vt,
                                                u16* __restrict__ Am,
                                                const float* __restrict__ lam_p,
                                                const float* __restrict__ ln_w) {
  __shared__ u16 Pl[16 * 40];  // P-tile transpose staging (padded)
  const int lane = threadIdx.x;
  const int bx = blockIdx.x;
  const int qt = bx & 127;
  const int pair = (bx >> 7) & 15;
  const int b = bx >> 11;
  const int q0 = qt * 16;
  const int lrw = lane & 15, lh = lane >> 4;
  const int khead = pair >> 1, vhead = pair >> 2;

  // Q fragments for both sub-heads (2i, 2i+1), d split 0..31 / 32..63
  bf16x16 qa[2][2];
  {
    const u16* qbase = Qm + ((size_t)(b * T_SEQ + q0 + lrw)) * EMBED;
    for (int p = 0; p < 2; p++) {
      int hc = (2 * pair + p) * HD;
      for (int d = 0; d < 2; d++) {
        const u16* pp = qbase + hc + d * 32 + lh * 8;
        qa[p][d] = frag2(pp, pp + 16);
      }
    }
  }

  f32x8 o[2][8] = {};            // 16 x 128 accumulator per sub-head
  float ml[2][8], ls[2][8];      // per-row running max / sum (rows j+8*lh)
  for (int p = 0; p < 2; p++)
    for (int j = 0; j < 8; j++) { ml[p][j] = -1e30f; ls[p][j] = 0.0f; }

  const float scale = 0.125f;  // 1/sqrt(64)
  const u16* kbase = Km + ((size_t)(b * T_SEQ)) * KV_COLS + khead * HD;
  const u16* vbase = Vt + ((size_t)(b * KV_COLS + vhead * 128 + lrw)) * T_SEQ;

  const int kend = q0 + 16;
  for (int kt = 0; kt < kend; kt += 32) {
    // K B-frags: lane = key col; per-lane d = lh*8 + {0..7} and +16
    bf16x16 kf[4];
    {
      const u16* k0p = kbase + (size_t)(kt + lrw) * KV_COLS + lh * 8;
      kf[0] = frag2(k0p, k0p + 16);         // keys kt..kt+15, d 0..31
      kf[1] = frag2(k0p + 32, k0p + 48);    // keys kt..kt+15, d 32..63
      const u16* k1p = k0p + (size_t)16 * KV_COLS;
      kf[2] = frag2(k1p, k1p + 16);         // keys kt+16..kt+31
      kf[3] = frag2(k1p + 32, k1p + 48);
    }
    if (kt + 32 < kend)
      __builtin_prefetch(kbase + (size_t)(kt + 32 + lrw) * KV_COLS, 0, 1);

    for (int p = 0; p < 2; p++) {
      f32x8 s0 = {}, s1 = {};
      s0 = wmma_bf16(qa[p][0], kf[0], s0);
      s0 = wmma_bf16(qa[p][1], kf[1], s0);
      s1 = wmma_bf16(qa[p][0], kf[2], s1);
      s1 = wmma_bf16(qa[p][1], kf[3], s1);

      const int c0 = kt + lrw, c1 = kt + 16 + lrw;
      float fac[8];
      for (int j = 0; j < 8; j++) {
        int r = q0 + j + 8 * lh;
        float a = s0[j] * scale; if (c0 > r) a = -1e30f;
        float c = s1[j] * scale; if (c1 > r) c = -1e30f;
        float mx = fmaxf(a, c);
        for (int m = 1; m < 16; m <<= 1) mx = fmaxf(mx, __shfl_xor(mx, m, 32));
        float mn = fmaxf(ml[p][j], mx);
        float f = __expf(ml[p][j] - mn);
        float e0 = __expf(a - mn), e1 = __expf(c - mn);
        float sum = e0 + e1;
        for (int m = 1; m < 16; m <<= 1) sum += __shfl_xor(sum, m, 32);
        ls[p][j] = ls[p][j] * f + sum;
        ml[p][j] = mn;
        fac[j] = f;
        // stash P (16 x 32 keys) into LDS, row-major, for A-frag reload
        int rr = j + 8 * lh;
        Pl[rr * 40 + lrw] = f2bf(e0);
        Pl[rr * 40 + 16 + lrw] = f2bf(e1);
      }
      for (int v = 0; v < 8; v++)
        for (int j = 0; j < 8; j++) o[p][v][j] *= fac[j];

      asm volatile("s_wait_dscnt 0" ::: "memory");
      bf16x16 pf;
      {
        const u16* pp = &Pl[lrw * 40 + lh * 8];  // A-frag pattern
        pf = frag2(pp, pp + 16);
      }
      asm volatile("" ::: "memory");

      for (int v = 0; v < 8; v++) {
        const u16* vp = vbase + (size_t)(v * 16) * T_SEQ + kt + lh * 8;
        bf16x16 vf = frag2(vp, vp + 16);
        o[p][v] = wmma_bf16(pf, vf, o[p][v]);
      }
    }
  }

  // -------- epilogue: attn1 - lambda*attn2, RMS norm, ln_w, (1-lambda_init)
  const float lam = lam_p[0];
  float lnw[8];
  for (int v = 0; v < 8; v++) lnw[v] = ln_w[v * 16 + lrw];
  float inv1[8], inv2[8], rst[8];
  for (int j = 0; j < 8; j++) {
    inv1[j] = 1.0f / ls[0][j];
    inv2[j] = 1.0f / ls[1][j];
    float s = 0.0f;
    for (int v = 0; v < 8; v++) {
      float x = o[0][v][j] * inv1[j] - lam * o[1][v][j] * inv2[j];
      s += x * x;
    }
    for (int m = 1; m < 16; m <<= 1) s += __shfl_xor(s, m, 32);
    rst[j] = rsqrtf(s * (1.0f / 128.0f) + 1e-5f);
  }
  const float oneml = 1.0f - LAMBDA_INIT_F;
  u16* obase = Am + ((size_t)(b * T_SEQ + q0)) * EMBED + pair * 128;
  for (int j = 0; j < 8; j++) {
    int r = j + 8 * lh;
    for (int v = 0; v < 8; v++) {
      float x = o[0][v][j] * inv1[j] - lam * o[1][v][j] * inv2[j];
      obase[(size_t)r * EMBED + v * 16 + lrw] = f2bf(x * rst[j] * lnw[v] * oneml);
    }
  }
}

// ---------------- host orchestration ----------------
extern "C" void kernel_launch(void* const* d_in, const int* in_sizes, int n_in,
                              void* d_out, int out_size, void* d_ws, size_t ws_size,
                              hipStream_t stream) {
  (void)in_sizes; (void)n_in; (void)out_size;
  const float* x   = (const float*)d_in[0];
  const float* Wq  = (const float*)d_in[1];
  const float* Wk  = (const float*)d_in[2];
  const float* Wv  = (const float*)d_in[3];
  const float* Wo  = (const float*)d_in[4];
  const float* lq1 = (const float*)d_in[5];
  const float* lk1 = (const float*)d_in[6];
  const float* lq2 = (const float*)d_in[7];
  const float* lk2 = (const float*)d_in[8];
  const float* lnw = (const float*)d_in[9];

  char* ws = (char*)d_ws;
  const size_t XB  = 0;                                  // x bf16   16 MB
  const size_t WQB = XB  + (size_t)BT * EMBED * 2;       // 8 MB
  const size_t WKB = WQB + (size_t)EMBED * EMBED * 2;    // 2 MB
  const size_t WVB = WKB + (size_t)KV_COLS * EMBED * 2;  // 2 MB
  const size_t WOB = WVB + (size_t)KV_COLS * EMBED * 2;  // 8 MB
  const size_t QB  = WOB + (size_t)EMBED * EMBED * 2;    // 16 MB
  const size_t KB  = QB  + (size_t)BT * EMBED * 2;       // 4 MB
  const size_t VB  = KB  + (size_t)BT * KV_COLS * 2;     // 4 MB
  const size_t VTB = VB  + (size_t)BT * KV_COLS * 2;     // 4 MB
  const size_t AB  = VTB + (size_t)BT * KV_COLS * 2;     // 16 MB
  const size_t LMB = AB  + (size_t)BT * EMBED * 2;
  const size_t NEED = LMB + 256;
  if (ws_size < NEED) return;

  u16* xb  = (u16*)(ws + XB);
  u16* wqb = (u16*)(ws + WQB);
  u16* wkb = (u16*)(ws + WKB);
  u16* wvb = (u16*)(ws + WVB);
  u16* wob = (u16*)(ws + WOB);
  u16* qb  = (u16*)(ws + QB);
  u16* kb  = (u16*)(ws + KB);
  u16* vb  = (u16*)(ws + VB);
  u16* vtb = (u16*)(ws + VTB);
  u16* ab  = (u16*)(ws + AB);
  float* lamb = (float*)(ws + LMB);

  // 1) convert to bf16
  auto cvt = [&](const float* src, u16* dst, int n) {
    cvt_f32_bf16<<<n / 1024, 256, 0, stream>>>(src, dst, n);
  };
  cvt(x,  xb,  BT * EMBED);
  cvt(Wq, wqb, EMBED * EMBED);
  cvt(Wk, wkb, KV_COLS * EMBED);
  cvt(Wv, wvb, KV_COLS * EMBED);
  cvt(Wo, wob, EMBED * EMBED);

  // 2) lambda_full
  lambda_kernel<<<1, 32, 0, stream>>>(lq1, lk1, lq2, lk2, lamb);

  // 3) projections (bf16 WMMA GEMMs, async double-buffered staging)
  gemm_xwT<false><<<dim3(EMBED / 128, BT / 128), 256, 0, stream>>>(
      xb, wqb, qb, BT, EMBED, EMBED);
  gemm_xwT<false><<<dim3(KV_COLS / 128, BT / 128), 256, 0, stream>>>(
      xb, wkb, kb, BT, KV_COLS, EMBED);
  gemm_xwT<false><<<dim3(KV_COLS / 128, BT / 128), 256, 0, stream>>>(
      xb, wvb, vb, BT, KV_COLS, EMBED);

  // 4) transpose V -> (b, vcol, t) for contiguous B-fragments
  transpose_v<<<dim3(KV_COLS / 32, T_SEQ / 32, BATCH), dim3(32, 8), 0, stream>>>(
      vb, vtb);

  // 5) differential flash attention + fused rmsnorm epilogue
  diff_attn<<<dim3(BATCH * NPAIR * (T_SEQ / 16)), 32, 0, stream>>>(
      qb, kb, vtb, ab, lamb, lnw);

  // 6) output projection (f32 out)
  gemm_xwT<true><<<dim3(EMBED / 128, BT / 128), 256, 0, stream>>>(
      ab, wob, d_out, BT, EMBED, EMBED);
}
